// CapsuleConv_4595615007178
// MI455X (gfx1250) — matrix operand order repaired and checked by reading, not compiled, not run on hardware
//
#include <hip/hip_runtime.h>
#include <hip/hip_bf16.h>

// ---------------------------------------------------------------------------
// CapsuleConv (first routing iter) on MI455X / gfx1250.
//
//   nv[b,m,w,a,d] = (1/32) * sum_{n,k,x} x[b,n,w+k,a*4+x] * w[k,n,x,d,m]
//   out = LayerNorm_16(nv)          (LayerNorm is scale-invariant -> drop 1/32)
//
// GEMM: [130944 x 384] x [384 x 128] in bf16 WMMA (v_wmma_f32_16x16x32_bf16),
// f32 accumulate. 8184 workgroups x (16 rows x 128 cols) tiles.
// A tile is staged into LDS directly in WMMA A-fragment layout so the inner
// loop is pure ds_load_b128 (immediate offsets) + v_wmma.
// ---------------------------------------------------------------------------

#define B_   16
#define N_   32
#define L_   2048
#define D_   16
#define M_   32
#define KW_  3
#define W_   2046           // output spatial length
#define NT_  12             // 384 / 32 K-steps per tile

typedef __attribute__((ext_vector_type(16))) __bf16 v16bf;
typedef __attribute__((ext_vector_type(8)))  float  v8f;

// f32 -> bf16, round-to-nearest-even
__device__ static inline unsigned short f32_bf16(float f) {
    union { float f; unsigned u; } c; c.f = f;
    unsigned u = c.u;
    u += 0x7FFFu + ((u >> 16) & 1u);
    return (unsigned short)(u >> 16);
}

// ---------------------------------------------------------------------------
// Pre-kernel: pack w[K][N][4][4][M] (f32) into bf16 B-fragments in d_ws.
// Fragment layout (16-bit B 32x16, wave32; contiguous K order, same as A):
//   lane l holds column n = l&15;  K = (l<16 ? 0 : 16) + e,  e = 0..15.
// wb[((t*8 + ctile)*32 + lane)*16 + e]
// ---------------------------------------------------------------------------
__global__ void caps_pack_w(const float* __restrict__ w,
                            unsigned short* __restrict__ wb) {
    int i = blockIdx.x * 256 + threadIdx.x;          // 0 .. 49151
    int e = i & 15;
    int l = (i >> 4) & 31;
    int c = (i >> 9) & 7;
    int t = i >> 12;                                  // 0..11
    int Kl  = ((l & 16) ? 16 : 0) + e;
    int kk  = 32 * t + Kl;                            // (n,k,x) flattened
    int n   = kk / 12;
    int rem = kk - n * 12;
    int k   = rem >> 2;
    int x   = rem & 3;
    int j   = 16 * c + (l & 15);                      // global column = m*4+d
    int m   = j >> 2;
    int d   = j & 3;
    float v = w[(((k * N_ + n) * 4 + x) * 4 + d) * M_ + m];
    wb[i] = f32_bf16(v);
}

// ---------------------------------------------------------------------------
// Main kernel: one WG = one 16-row x 128-col tile. 8 waves; wave c owns
// column tile c. Row r = ((b*W + w)*4 + a); tile T covers groups 4T..4T+3.
// ---------------------------------------------------------------------------
__launch_bounds__(256)
__global__ void caps_main(const float* __restrict__ xin,
                          const unsigned short* __restrict__ wb,
                          const float* __restrict__ gamma,
                          const float* __restrict__ beta,
                          float* __restrict__ out) {
    // A fragments, WMMA-ready: Afr[t][lane][e], e=0..15 bf16  (12 KB)
    __shared__ unsigned short Afr[NT_ * 32 * 16];
    __shared__ float C[16 * 132];               // f32 result tile, padded

    const int tid  = threadIdx.x;
    const int lane = tid & 31;
    const int wave = tid >> 5;
    const int T    = blockIdx.x;

    // ---- B fragments for this wave's column tile: 12 x v16bf in registers
    const v16bf* wbv = (const v16bf*)wb;
    v16bf bfrag[NT_];
#pragma unroll
    for (int t = 0; t < NT_; ++t)
        bfrag[t] = wbv[(t * 8 + wave) * 32 + lane];

    // ---- cooperative A staging: coalesced float4 loads of x[b,n,w+k,a*4..+3],
    // bf16 convert, scattered directly into fragment layout.
    // One float4 == one 4-wide K-chunk (kk = n*12+k*4 .. +3) of one fragment:
    //   t = kkb>>5, ch = (kkb>>2)&7, hi = ch>>2, vp = ch&3,
    //   fragment lane = hi*16 + gl*4 + a, elements vp*4 .. vp*4+3.
#pragma unroll
    for (int j = 0; j < 6; ++j) {
        int i  = tid + j * 256;                 // 0 .. 1535
        int gl = i / 384;                       // group in tile (row / 4)
        int r  = i - gl * 384;
        int n  = r / 12;
        int q  = r - n * 12;
        int k  = q >> 2;
        int a  = q & 3;
        int g  = 4 * T + gl;
        int b  = g / W_;
        int w  = g - b * W_;
        const float4 v =
            *(const float4*)(xin + (((b * N_ + n) * L_ + (w + k)) << 4) + a * 4);
        ushort4 h;
        h.x = f32_bf16(v.x); h.y = f32_bf16(v.y);
        h.z = f32_bf16(v.z); h.w = f32_bf16(v.w);

        int kkb   = n * 12 + k * 4;
        int t     = kkb >> 5;
        int ch    = (kkb >> 2) & 7;
        int hi    = ch >> 2;
        int vp    = ch & 3;
        int lfr   = hi * 16 + gl * 4 + a;
        *(ushort4*)&Afr[(t * 32 + lfr) * 16 + vp * 4] = h;
    }
    __syncthreads();

    // ---- GEMM: 12 K-steps, pure ds_load_b128 + v_wmma, dual accumulators
    const unsigned short* aptr = &Afr[lane * 16];
    v8f acc0 = {0.f, 0.f, 0.f, 0.f, 0.f, 0.f, 0.f, 0.f};
    v8f acc1 = {0.f, 0.f, 0.f, 0.f, 0.f, 0.f, 0.f, 0.f};
#pragma unroll
    for (int t = 0; t < NT_; ++t) {
        const v16bf afr = *(const v16bf*)(aptr + t * (32 * 16));
        if (t & 1)
            acc1 = __builtin_amdgcn_wmma_f32_16x16x32_bf16(
                false, afr, false, bfrag[t], (short)0, acc1, false, false);
        else
            acc0 = __builtin_amdgcn_wmma_f32_16x16x32_bf16(
                false, afr, false, bfrag[t], (short)0, acc0, false, false);
    }
    v8f acc = acc0 + acc1;

    // ---- spill 16x16 tile to LDS (C/D layout: VGPR q -> row q (+8 hi half))
    const int rl   = lane & 15;
    const int colb = wave * 16 + rl;
    const int rowb = (lane < 16) ? 0 : 8;
#pragma unroll
    for (int q = 0; q < 8; ++q)
        C[(rowb + q) * 132 + colb] = acc[q];
    __syncthreads();

    // ---- LayerNorm: 128 groups of 16 = (4 rows a) x (4 cols d) sub-blocks
    if (tid < 128) {
        int gl = tid >> 5;                       // group-in-tile
        int m  = tid & 31;                       // out capsule
        float vals[16];
#pragma unroll
        for (int a = 0; a < 4; ++a) {
            float4 v = *(const float4*)&C[(gl * 4 + a) * 132 + m * 4];
            vals[a * 4 + 0] = v.x; vals[a * 4 + 1] = v.y;
            vals[a * 4 + 2] = v.z; vals[a * 4 + 3] = v.w;
        }
        float mu = 0.f;
#pragma unroll
        for (int i = 0; i < 16; ++i) mu += vals[i];
        mu *= (1.f / 16.f);
        float var = 0.f;
#pragma unroll
        for (int i = 0; i < 16; ++i) {
            float d = vals[i] - mu;
            var += d * d;
        }
        var *= (1.f / 16.f);
        const float inv = rsqrtf(var + 1e-5f);

        int g = 4 * T + gl;
        int b = g / W_;
        int w = g - b * W_;
        float* op = out + (((b * M_ + m) * W_ + w) << 4);
#pragma unroll
        for (int a = 0; a < 4; ++a) {
            float4 o;
            o.x = gamma[a*4+0] * (vals[a*4+0] - mu) * inv + beta[a*4+0];
            o.y = gamma[a*4+1] * (vals[a*4+1] - mu) * inv + beta[a*4+1];
            o.z = gamma[a*4+2] * (vals[a*4+2] - mu) * inv + beta[a*4+2];
            o.w = gamma[a*4+3] * (vals[a*4+3] - mu) * inv + beta[a*4+3];
            *(float4*)(op + a * 4) = o;
        }
    }
}

extern "C" void kernel_launch(void* const* d_in, const int* in_sizes, int n_in,
                              void* d_out, int out_size, void* d_ws, size_t ws_size,
                              hipStream_t stream) {
    const float* x     = (const float*)d_in[0];
    const float* w     = (const float*)d_in[1];
    const float* gamma = (const float*)d_in[2];
    const float* beta  = (const float*)d_in[3];
    // d_in[4] = num_iter (unused on first routing pass)

    unsigned short* wb = (unsigned short*)d_ws;   // 96 KB bf16 B-fragments

    caps_pack_w<<<192, 256, 0, stream>>>(w, wb);                  // 49152 elems
    caps_main<<<(B_ * W_) / 4, 256, 0, stream>>>(x, wb, gamma, beta,
                                                 (float*)d_out);  // 8184 WGs
}
